// TiFeAttention_24292335026587
// MI455X (gfx1250) — matrix operand order repaired and checked by reading, not compiled, not run on hardware
//
#include <hip/hip_runtime.h>
#include <hip/hip_bf16.h>
#include <stdint.h>

// ---------------------------------------------------------------------------
// TiFeAttention for MI455X (gfx1250, wave32, WMMA).
//   * One-time pass converts x and all weights to bf16; weights are stored
//     pre-transposed into the [n][k] layout the WMMA B-fragment wants.
//   * GEMM: bf16 A/B, f32 WMMA accumulation; tiles staged with
//     GLOBAL_LOAD_ASYNC_TO_LDS_B128 into double-buffered LDS (ASYNCcnt).
//   * GEMM epilogue writes f32 and/or bf16 (optionally transposed) so every
//     consumer reads its operand in native layout; no guards (all dims are
//     multiples of the 128x128x32 tiling).
// ---------------------------------------------------------------------------

typedef __attribute__((ext_vector_type(8)))  float          v8f;
typedef __attribute__((ext_vector_type(16))) __bf16         bf16x16;
typedef __attribute__((ext_vector_type(8)))  unsigned short ushort8;
typedef __attribute__((ext_vector_type(4)))  int            int4v;
typedef long long ll;

#define LDS_STRIDE 40  // 32 K-elems + pad; 80B row pitch keeps 16B alignment

__device__ __forceinline__ unsigned short f32_to_bf16_rne(float f) {
  unsigned int u = __float_as_uint(f);
  u += 0x7FFFu + ((u >> 16) & 1u);
  return (unsigned short)(u >> 16);
}

// -------------------- async global->LDS plumbing ---------------------------
#if __has_builtin(__builtin_amdgcn_global_load_async_to_lds_b32) && \
    __has_builtin(__builtin_amdgcn_s_wait_asynccnt)
#define HAVE_ASYNC32 1
#else
#define HAVE_ASYNC32 0
#endif
#if __has_builtin(__builtin_amdgcn_global_load_async_to_lds_b128) && \
    __has_builtin(__builtin_amdgcn_s_wait_asynccnt)
#define HAVE_ASYNC128 1
#else
#define HAVE_ASYNC128 0
#endif

typedef __attribute__((address_space(1))) int   gint_t;
typedef __attribute__((address_space(3))) int   lint_t;
typedef __attribute__((address_space(1))) int4v gint4_t;
typedef __attribute__((address_space(3))) int4v lint4_t;

__device__ __forceinline__ gint_t* as_global(const void* p) {
  return (gint_t*)(unsigned long long)(uintptr_t)p;
}
__device__ __forceinline__ lint_t* as_lds(const void* p) {
  return (lint_t*)(unsigned int)(uintptr_t)p;
}
__device__ __forceinline__ gint4_t* as_global4(const void* p) {
  return (gint4_t*)(unsigned long long)(uintptr_t)p;
}
__device__ __forceinline__ lint4_t* as_lds4(const void* p) {
  return (lint4_t*)(unsigned int)(uintptr_t)p;
}

// Copy 16 bytes global -> LDS (addresses 16B aligned).
__device__ __forceinline__ void async_copy16(const unsigned short* g,
                                             unsigned short* l) {
#if HAVE_ASYNC128
  __builtin_amdgcn_global_load_async_to_lds_b128(as_global4(g), as_lds4(l), 0, 0);
#elif HAVE_ASYNC32
  __builtin_amdgcn_global_load_async_to_lds_b32(as_global(g + 0), as_lds(l + 0), 0, 0);
  __builtin_amdgcn_global_load_async_to_lds_b32(as_global(g + 2), as_lds(l + 2), 0, 0);
  __builtin_amdgcn_global_load_async_to_lds_b32(as_global(g + 4), as_lds(l + 4), 0, 0);
  __builtin_amdgcn_global_load_async_to_lds_b32(as_global(g + 6), as_lds(l + 6), 0, 0);
#else
  *(ushort8*)l = *(const ushort8*)g;
#endif
}

__device__ __forceinline__ void wait_async_zero() {
#if HAVE_ASYNC32 || HAVE_ASYNC128
  __builtin_amdgcn_s_wait_asynccnt(0);
#endif
}

// -------------------- WMMA fragment loads ----------------------------------
union Frag16 { ushort8 h[2]; bf16x16 v; };

// A fragment, 16x32 bf16 (MxK): lanes hold M = lane&15,
// K chunks {koff..koff+7, koff+16..koff+23}, koff = (lane>>4)*8.
__device__ __forceinline__ bf16x16 load_a_frag(const unsigned short* sA,
                                               int mBase, int lane) {
  int row  = mBase + (lane & 15);
  int koff = (lane >> 4) << 3;
  Frag16 f;
  f.h[0] = *(const ushort8*)(sA + row * LDS_STRIDE + koff);
  f.h[1] = *(const ushort8*)(sA + row * LDS_STRIDE + koff + 16);
  return f.v;
}

// B fragment, 32x16 bf16 (KxN), LDS tile stored [n][k]: lane holds
// N = nBase + (lane&15), K chunks {koff..koff+7, koff+16..koff+23}, koff = (lane>>4)*8.
__device__ __forceinline__ bf16x16 load_b_frag(const unsigned short* sB,
                                               int nBase, int lane) {
  int col  = nBase + (lane & 15);
  int koff = (lane >> 4) << 3;
  Frag16 f;
  f.h[0] = *(const ushort8*)(sB + col * LDS_STRIDE + koff);
  f.h[1] = *(const ushort8*)(sB + col * LDS_STRIDE + koff + 16);
  return f.v;
}

// ---------------------------------------------------------------------------
// GEMM: C/Cb = act(alpha * A @ B^T_layout + bias).
//   A  : bf16, row-major M x K (lda)
//   Bm : bf16, [n][k] layout N x K (ldb)   (i.e. B already transposed)
//   C  : optional f32 out (ldc);  Cb: optional bf16 out (ldcb, opt transposed)
// Workgroup tile 128x128, 256 threads = 8 waves (2 x 4), wave tile 64x32.
// K-tiles of 32 staged via async DMA into double-buffered LDS.
// Requires: M,N multiples of 128; K multiple of 32 (true for every launch).
// ---------------------------------------------------------------------------
__global__ __launch_bounds__(256) void gemm_bf16_wmma(
    const unsigned short* __restrict__ A, const unsigned short* __restrict__ Bm,
    const float* __restrict__ bias, float* __restrict__ C,
    unsigned short* __restrict__ Cb, int cbTrans, int M, int N, int K, ll lda,
    ll ldb, ll ldc, ll ldcb, ll strideA, ll strideB, ll strideC, ll strideCb,
    float alpha, int doRelu) {
  __shared__ unsigned short sA[2][128 * LDS_STRIDE];
  __shared__ unsigned short sB[2][128 * LDS_STRIDE];

  const int z = blockIdx.z;
  A += (ll)z * strideA;
  Bm += (ll)z * strideB;
  if (C) C += (ll)z * strideC;
  if (Cb) Cb += (ll)z * strideCb;

  const int tileM = blockIdx.y * 128;
  const int tileN = blockIdx.x * 128;
  const int tid = threadIdx.x;
  const int wave = tid >> 5;
  const int lane = tid & 31;
  const int waveM = (wave >> 2) * 64;  // 2 waves down M
  const int waveN = (wave & 3) * 32;   // 4 waves across N

  // Stage one 128x32 bf16 tile (8KB): 512 16B-chunks over 256 threads.
  auto stageA = [&](int k0, int buf) {
#pragma unroll
    for (int it = 0; it < 2; ++it) {
      int idx = tid + it * 256;
      int row = idx >> 2, ch = idx & 3;
      async_copy16(A + (ll)(tileM + row) * lda + k0 + ch * 8,
                   &sA[buf][row * LDS_STRIDE + ch * 8]);
    }
  };
  auto stageB = [&](int k0, int buf) {
#pragma unroll
    for (int it = 0; it < 2; ++it) {
      int idx = tid + it * 256;
      int row = idx >> 2, ch = idx & 3;
      async_copy16(Bm + (ll)(tileN + row) * ldb + k0 + ch * 8,
                   &sB[buf][row * LDS_STRIDE + ch * 8]);
    }
  };

  v8f acc[4][2] = {};
  const int nk = K >> 5;

  stageA(0, 0);
  stageB(0, 0);
  for (int t = 0; t < nk; ++t) {
    const int cur = t & 1;
    wait_async_zero();   // my DMA into sX[cur] landed
    __syncthreads();     // everyone's did
    if (t + 1 < nk) {    // fire next tile's DMA while we compute
      stageA((t + 1) << 5, cur ^ 1);
      stageB((t + 1) << 5, cur ^ 1);
    }
    bf16x16 aF[4], bF[2];
#pragma unroll
    for (int i = 0; i < 4; ++i) aF[i] = load_a_frag(sA[cur], waveM + 16 * i, lane);
#pragma unroll
    for (int j = 0; j < 2; ++j) bF[j] = load_b_frag(sB[cur], waveN + 16 * j, lane);
#pragma unroll
    for (int i = 0; i < 4; ++i)
#pragma unroll
      for (int j = 0; j < 2; ++j)
        acc[i][j] = __builtin_amdgcn_wmma_f32_16x16x32_bf16(
            false, aF[i], false, bF[j], (short)0, acc[i][j], false, false);
  }

  // Epilogue. C layout: VGPR r -> lanes 0-15: M=r, lanes 16-31: M=r+8.
  const int halfM = (lane >> 4) * 8;
  const int cn = lane & 15;
#pragma unroll
  for (int i = 0; i < 4; ++i)
#pragma unroll
    for (int j = 0; j < 2; ++j) {
      const int col = tileN + waveN + 16 * j + cn;
#pragma unroll
      for (int r = 0; r < 8; ++r) {
        const int row = tileM + waveM + 16 * i + r + halfM;
        float v = acc[i][j][r] * alpha;
        if (bias) v += bias[col];
        if (doRelu) v = v > 0.0f ? v : 0.0f;
        if (C) C[(ll)row * ldc + col] = v;
        if (Cb) {
          unsigned short h = f32_to_bf16_rne(v);
          if (cbTrans) Cb[(ll)col * ldcb + row] = h;
          else         Cb[(ll)row * ldcb + col] = h;
        }
      }
    }
}

// ---------------------------------------------------------------------------
// Row softmax, rows of 256 f32 in -> 256 bf16 out (scale folded into GEMM).
// Row load uses the async global->LDS path when available.
// ---------------------------------------------------------------------------
__global__ __launch_bounds__(256) void softmax_rows_256(
    const float* __restrict__ src, unsigned short* __restrict__ dst) {
  __shared__ float sv[256];
  __shared__ float red[256];
  const ll row = blockIdx.x;
  const float* rowp = src + row * 256;
  const int t = threadIdx.x;
#if HAVE_ASYNC32
  __builtin_amdgcn_global_load_async_to_lds_b32(as_global(rowp + t),
                                                as_lds(&sv[t]), 0, 0);
  __builtin_amdgcn_s_wait_asynccnt(0);
  __syncthreads();
  float v = sv[t];
#else
  float v = rowp[t];
#endif
  red[t] = v;
  __syncthreads();
  for (int s = 128; s > 0; s >>= 1) {
    if (t < s) red[t] = fmaxf(red[t], red[t + s]);
    __syncthreads();
  }
  float m = red[0];
  __syncthreads();
  float e = __expf(v - m);
  red[t] = e;
  __syncthreads();
  for (int s = 128; s > 0; s >>= 1) {
    if (t < s) red[t] += red[t + s];
    __syncthreads();
  }
  dst[row * 256 + t] = f32_to_bf16_rne(e * (1.0f / red[0]));
}

// f32 -> bf16 elementwise
__global__ __launch_bounds__(256) void convert_bf16(
    const float* __restrict__ src, unsigned short* __restrict__ dst, ll n) {
  ll i = (ll)blockIdx.x * 256 + threadIdx.x;
  if (i < n) dst[i] = f32_to_bf16_rne(src[i]);
}

// dst (Cc x R) = transpose(src (R x Cc)), f32 -> bf16, batched over z.
__global__ __launch_bounds__(256) void transpose_convert_bf16(
    const float* __restrict__ src, unsigned short* __restrict__ dst, int R,
    int Cc, ll strideSrc, ll strideDst) {
  __shared__ float tile[16][17];
  const int z = blockIdx.z;
  src += (ll)z * strideSrc;
  dst += (ll)z * strideDst;
  const int r0 = blockIdx.y * 16, c0 = blockIdx.x * 16;
  const int tx = threadIdx.x & 15, ty = threadIdx.x >> 4;
  tile[ty][tx] = src[(ll)(r0 + ty) * Cc + (c0 + tx)];
  __syncthreads();
  dst[(ll)(c0 + ty) * R + (r0 + tx)] = f32_to_bf16_rne(tile[tx][ty]);
}

// ann[r] = sum_j H[r, j] * w[j] + b[0]
__global__ __launch_bounds__(256) void rowdot_256(const float* __restrict__ H,
                                                  const float* __restrict__ w,
                                                  const float* __restrict__ b,
                                                  float* __restrict__ out) {
  __shared__ float red[256];
  const ll r = blockIdx.x;
  const int t = threadIdx.x;
  red[t] = H[r * 256 + t] * w[t];
  __syncthreads();
  for (int s = 128; s > 0; s >>= 1) {
    if (t < s) red[t] += red[t + s];
    __syncthreads();
  }
  if (t == 0) out[r] = red[0] + b[0];
}

// out[b, t, n] = ann[b*512 + t] * ann[b*512 + 256 + n]
__global__ __launch_bounds__(256) void outer_bt_bn(const float* __restrict__ ann,
                                                   float* __restrict__ out) {
  const int b = blockIdx.y, tt = blockIdx.x, n = threadIdx.x;
  out[((ll)b * 256 + tt) * 256 + n] =
      ann[b * 512 + tt] * ann[b * 512 + 256 + n];
}

// ---------------------------------------------------------------------------
typedef unsigned short bf16_t;

static void launch_gemm(const bf16_t* A, const bf16_t* Bm, const float* bias,
                        float* C, bf16_t* Cb, int cbTrans, int M, int N, int K,
                        ll lda, ll ldb, ll ldc, ll ldcb, ll sA, ll sB, ll sC,
                        ll sCb, float alpha, int relu, int batch,
                        hipStream_t s) {
  dim3 grid((unsigned)(N / 128), (unsigned)(M / 128), (unsigned)batch);
  gemm_bf16_wmma<<<grid, dim3(256), 0, s>>>(A, Bm, bias, C, Cb, cbTrans, M, N,
                                            K, lda, ldb, ldc, ldcb, sA, sB, sC,
                                            sCb, alpha, relu);
}

extern "C" void kernel_launch(void* const* d_in, const int* in_sizes, int n_in,
                              void* d_out, int out_size, void* d_ws,
                              size_t ws_size, hipStream_t stream) {
  (void)in_sizes; (void)n_in; (void)out_size; (void)ws_size;
  const float* x    = (const float*)d_in[0];
  const float* tq_w = (const float*)d_in[1];  const float* tq_b = (const float*)d_in[2];
  const float* tk_w = (const float*)d_in[3];  const float* tk_b = (const float*)d_in[4];
  const float* tv_w = (const float*)d_in[5];  const float* tv_b = (const float*)d_in[6];
  const float* fq_w = (const float*)d_in[7];  const float* fq_b = (const float*)d_in[8];
  const float* fk_w = (const float*)d_in[9];  const float* fk_b = (const float*)d_in[10];
  const float* fv_w = (const float*)d_in[11]; const float* fv_b = (const float*)d_in[12];
  const float* wq_w = (const float*)d_in[13]; const float* wq_b = (const float*)d_in[14];
  const float* wk_w = (const float*)d_in[15]; const float* wk_b = (const float*)d_in[16];
  const float* wv_w = (const float*)d_in[17]; const float* wv_b = (const float*)d_in[18];
  const float* wo_w = (const float*)d_in[19]; const float* wo_b = (const float*)d_in[20];
  const float* e1_w = (const float*)d_in[21]; const float* e1_b = (const float*)d_in[22];
  const float* e2_w = (const float*)d_in[23]; const float* e2_b = (const float*)d_in[24];

  const int Bsz = 256;
  const ll PM = 65536;              // 256*256
  const ll FF = 131072;             // (T+N)*D
  const float scale = 1.0f / 16.0f;

  // ---- workspace carve (256B aligned) ----
  char* p = (char*)d_ws;
  auto alloc = [&](size_t bytes) {
    void* r = (void*)p;
    p += (bytes + 255) & ~(size_t)255;
    return r;
  };
  bf16_t* xbf   = (bf16_t*)alloc(Bsz * PM * 2);        // x,       (B,T,N)
  bf16_t* xbfT  = (bf16_t*)alloc(Bsz * PM * 2);        // x^T,     (B,N,T)
  bf16_t* wTtq  = (bf16_t*)alloc(PM * 2);
  bf16_t* wTtk  = (bf16_t*)alloc(PM * 2);
  bf16_t* wTtv  = (bf16_t*)alloc(PM * 2);
  bf16_t* wTfq  = (bf16_t*)alloc(PM * 2);
  bf16_t* wTfk  = (bf16_t*)alloc(PM * 2);
  bf16_t* wTfv  = (bf16_t*)alloc(PM * 2);
  bf16_t* wTe1  = (bf16_t*)alloc(PM * 2);
  bf16_t* wTwq  = (bf16_t*)alloc(256 * FF * 2);        // (D,F) = wq_w^T
  bf16_t* wTwk  = (bf16_t*)alloc(256 * FF * 2);
  bf16_t* wTwv  = (bf16_t*)alloc(256 * FF * 2);
  bf16_t* wTwo  = (bf16_t*)alloc(FF * 256 * 2);        // (F,D) = wo_w^T
  bf16_t* QB    = (bf16_t*)alloc(Bsz * PM * 2);
  bf16_t* KB    = (bf16_t*)alloc(Bsz * PM * 2);
  bf16_t* VBT   = (bf16_t*)alloc(Bsz * PM * 2);        // V^T per batch
  bf16_t* atB   = (bf16_t*)alloc(Bsz * PM * 2);
  float*  Sc    = (float*)alloc(Bsz * PM * 4);         // raw scores (f32)
  bf16_t* catB  = (bf16_t*)alloc((size_t)Bsz * FF * 2);// (B,512,256)
  bf16_t* QwB   = (bf16_t*)alloc(PM * 2);
  bf16_t* KwB   = (bf16_t*)alloc(PM * 2);
  bf16_t* VwTB  = (bf16_t*)alloc(PM * 2);
  float*  AwF   = (float*)alloc(PM * 4);
  bf16_t* AwB   = (bf16_t*)alloc(PM * 2);
  bf16_t* WattB = (bf16_t*)alloc(PM * 2);
  bf16_t* woutB = (bf16_t*)alloc((size_t)Bsz * FF * 2);
  float*  Hbuf  = (float*)alloc((size_t)Bsz * 512 * 256 * 4);
  float*  Ann   = (float*)alloc((size_t)Bsz * 512 * 4);
  float*  out   = (float*)d_out;

  // ---- one-time bf16 conversion / weight transposition ----
  convert_bf16<<<dim3(65536), dim3(256), 0, stream>>>(x, xbf, (ll)Bsz * PM);
  transpose_convert_bf16<<<dim3(16, 16, Bsz), dim3(256), 0, stream>>>(x, xbfT, 256, 256, PM, PM);
  transpose_convert_bf16<<<dim3(16, 16, 1), dim3(256), 0, stream>>>(tq_w, wTtq, 256, 256, 0, 0);
  transpose_convert_bf16<<<dim3(16, 16, 1), dim3(256), 0, stream>>>(tk_w, wTtk, 256, 256, 0, 0);
  transpose_convert_bf16<<<dim3(16, 16, 1), dim3(256), 0, stream>>>(tv_w, wTtv, 256, 256, 0, 0);
  transpose_convert_bf16<<<dim3(16, 16, 1), dim3(256), 0, stream>>>(fq_w, wTfq, 256, 256, 0, 0);
  transpose_convert_bf16<<<dim3(16, 16, 1), dim3(256), 0, stream>>>(fk_w, wTfk, 256, 256, 0, 0);
  transpose_convert_bf16<<<dim3(16, 16, 1), dim3(256), 0, stream>>>(fv_w, wTfv, 256, 256, 0, 0);
  transpose_convert_bf16<<<dim3(16, 16, 1), dim3(256), 0, stream>>>(e1_w, wTe1, 256, 256, 0, 0);
  transpose_convert_bf16<<<dim3(16, 8192, 1), dim3(256), 0, stream>>>(wq_w, wTwq, 131072, 256, 0, 0);
  transpose_convert_bf16<<<dim3(16, 8192, 1), dim3(256), 0, stream>>>(wk_w, wTwk, 131072, 256, 0, 0);
  transpose_convert_bf16<<<dim3(16, 8192, 1), dim3(256), 0, stream>>>(wv_w, wTwv, 131072, 256, 0, 0);
  transpose_convert_bf16<<<dim3(8192, 16, 1), dim3(256), 0, stream>>>(wo_w, wTwo, 256, 131072, 0, 0);

  // ---- Time attention ----
  launch_gemm(xbf, wTtq, tq_b, nullptr, QB, 0, 256, 256, 256, 256, 256, 0, 256, PM, 0, 0, PM, 1.0f, 0, Bsz, stream);
  launch_gemm(xbf, wTtk, tk_b, nullptr, KB, 0, 256, 256, 256, 256, 256, 0, 256, PM, 0, 0, PM, 1.0f, 0, Bsz, stream);
  launch_gemm(xbf, wTtv, tv_b, nullptr, VBT, 1, 256, 256, 256, 256, 256, 0, 256, PM, 0, 0, PM, 1.0f, 0, Bsz, stream);
  launch_gemm(QB, KB, nullptr, Sc, nullptr, 0, 256, 256, 256, 256, 256, 256, 0, PM, PM, PM, 0, scale, 0, Bsz, stream);
  softmax_rows_256<<<dim3(65536), dim3(256), 0, stream>>>(Sc, atB);
  launch_gemm(atB, VBT, nullptr, nullptr, catB, 0, 256, 256, 256, 256, 256, 0, 256, PM, PM, 0, FF, 1.0f, 0, Bsz, stream);

  // ---- Feature attention (A = x^T) ----
  launch_gemm(xbfT, wTfq, fq_b, nullptr, QB, 0, 256, 256, 256, 256, 256, 0, 256, PM, 0, 0, PM, 1.0f, 0, Bsz, stream);
  launch_gemm(xbfT, wTfk, fk_b, nullptr, KB, 0, 256, 256, 256, 256, 256, 0, 256, PM, 0, 0, PM, 1.0f, 0, Bsz, stream);
  launch_gemm(xbfT, wTfv, fv_b, nullptr, VBT, 1, 256, 256, 256, 256, 256, 0, 256, PM, 0, 0, PM, 1.0f, 0, Bsz, stream);
  launch_gemm(QB, KB, nullptr, Sc, nullptr, 0, 256, 256, 256, 256, 256, 256, 0, PM, PM, PM, 0, scale, 0, Bsz, stream);
  softmax_rows_256<<<dim3(65536), dim3(256), 0, stream>>>(Sc, atB);
  // feat_att written transposed straight into cat's second half:
  // cat[b, 256+d, n] = feat_att[b, n, d]
  launch_gemm(atB, VBT, nullptr, nullptr, catB + PM, 1, 256, 256, 256, 256, 256, 0, 256, PM, PM, 0, FF, 1.0f, 0, Bsz, stream);

  // ---- Sample (batch-dim) attention over flat = cat.reshape(B, 131072) ----
  launch_gemm(catB, wTwq, wq_b, nullptr, QwB, 0, 256, 256, 131072, 131072, 131072, 0, 256, 0, 0, 0, 0, 1.0f, 0, 1, stream);
  launch_gemm(catB, wTwk, wk_b, nullptr, KwB, 0, 256, 256, 131072, 131072, 131072, 0, 256, 0, 0, 0, 0, 1.0f, 0, 1, stream);
  launch_gemm(catB, wTwv, wv_b, nullptr, VwTB, 1, 256, 256, 131072, 131072, 131072, 0, 256, 0, 0, 0, 0, 1.0f, 0, 1, stream);
  launch_gemm(QwB, KwB, nullptr, AwF, nullptr, 0, 256, 256, 256, 256, 256, 256, 0, 0, 0, 0, 0, scale, 0, 1, stream);
  softmax_rows_256<<<dim3(256), dim3(256), 0, stream>>>(AwF, AwB);
  launch_gemm(AwB, VwTB, nullptr, nullptr, WattB, 0, 256, 256, 256, 256, 256, 0, 256, 0, 0, 0, 0, 1.0f, 0, 1, stream);
  launch_gemm(WattB, wTwo, wo_b, nullptr, woutB, 0, 256, 131072, 256, 256, 256, 0, 131072, 0, 0, 0, 0, 1.0f, 0, 1, stream);

  // ---- MLP head ----
  launch_gemm(woutB, wTe1, e1_b, Hbuf, nullptr, 0, 131072, 256, 256, 256, 256, 256, 0, 0, 0, 0, 0, 1.0f, 1, 1, stream);
  rowdot_256<<<dim3(131072), dim3(256), 0, stream>>>(Hbuf, e2_w, e2_b, Ann);
  outer_bt_bn<<<dim3(256, 256), dim3(256), 0, stream>>>(Ann, out);
}
